// B2BsqrtTANDEM_3590592660259
// MI455X (gfx1250) — compile-verified, matrix operand order checked by reading, not run on hardware
//
#include <hip/hip_runtime.h>

// ---------------------------------------------------------------------------
// CDNA5 (gfx1250) LSTM-b2bsqrt, WMMA bf16 path + double-buffered TDM staging.
//   B=512, T=100, F=H=1024, gates=4 [i,f,o,c], C=10
// Phases:
//   0. transpose+convert W,U -> bf16 [g][n][k] (k-contiguous B operand =>
//      all LDS fragment reads are ds_load_b128); x -> bf16
//   1. xz[m][g*H+h] = x[m,:] @ W[g,:,h] + bias   (M=51200, N=4096, K=1024)
//   2. per step t: z = xz_t + h @ U ; gate math ; c,h update (4 gates/wave)
//   3. LayerNorm + FC(1024->10)
// Global->LDS tile staging: TENSOR_LOAD_TO_LDS, software-pipelined with two
// LDS buffers so the DMA for chunk k+1 runs under the WMMAs of chunk k.
// ---------------------------------------------------------------------------

typedef __attribute__((ext_vector_type(16))) __bf16 v16bf;
typedef __attribute__((ext_vector_type(8)))  float  v8f;
typedef __attribute__((ext_vector_type(4)))  unsigned int v4u;
typedef __attribute__((ext_vector_type(8)))  int  v8i_;
typedef __attribute__((ext_vector_type(4)))  int  v4i_;

#define B_   512
#define T_   100
#define K_   1024
#define H_   1024
#define N4_  4096   // 4 gates * H

#if __has_builtin(__builtin_amdgcn_tensor_load_to_lds)
#define USE_TDM 1
#else
#define USE_TDM 0
#endif

__device__ __forceinline__ __bf16 f2bf(float f) {
  unsigned u = __builtin_bit_cast(unsigned, f);
  u += 0x7FFFu + ((u >> 16) & 1u);          // round to nearest even
  unsigned short h = (unsigned short)(u >> 16);
  return __builtin_bit_cast(__bf16, h);
}

__device__ __forceinline__ float b2bsqrt_f(float x) {
  float s = (x >= 0.f) ? 1.f : -1.f;
  return s * (sqrtf(1.f + fabsf(x)) - 1.f);
}

__device__ __forceinline__ float sigmoid_f(float z) {
  return 1.f / (1.f + __expf(-z));
}

#if USE_TDM
// TDM 2D tile load: `rows` x 32 bf16 tile, row-contiguous in global with a
// 1024-element row stride; LDS dest gets 8 bf16 (4 DWORD) padding after every
// 32-element (16 DWORD) row -> matches __shared__ [rows][32+8] layout.
__device__ __forceinline__ void tdm_load_2d(unsigned lds_addr, const void* gptr,
                                            unsigned rows) {
  unsigned long long ga = (unsigned long long)(uintptr_t)gptr;
  v4u g0;
  g0[0] = 1u;                                             // count=1, no gather
  g0[1] = lds_addr;                                       // lds_addr[31:0]
  g0[2] = (unsigned)(ga & 0xFFFFFFFFu);                   // global_addr[31:0]
  g0[3] = (unsigned)((ga >> 32) & 0x1FFFFFFu) | (2u << 30); // ga[56:32] | type=2
  v8i_ g1;
  g1[0] = (int)((1u << 16)    // data_size = 2 bytes
              | (1u << 20)    // pad_enable
              | (3u << 22)    // pad_interval: every 16 DWORDs (64B row)
              | (3u << 25));  // pad_amount: 4 DWORDs (8 bf16)
  g1[1] = (int)(1024u << 16);          // tensor_dim0[15:0]=1024 (atomic addr=0)
  g1[2] = (int)(1024u << 16);          // tensor_dim0[31:16]=0 | tensor_dim1[15:0]=1024
  g1[3] = (int)(32u << 16);            // tensor_dim1[31:16]=0 | tile_dim0=32
  g1[4] = (int)(rows & 0xFFFFu);       // tile_dim1=rows, tile_dim2=0
  g1[5] = (int)1024;                   // tensor_dim0_stride[31:0] = 1024 elems
  g1[6] = 0;                           // stride0[47:32]=0 | stride1[15:0]=0
  g1[7] = 0;                           // stride1[47:16]=0
  v4i_ z = {0, 0, 0, 0};
#if defined(__clang_major__) && __clang_major__ >= 23
  v8i_ z8 = {};
  __builtin_amdgcn_tensor_load_to_lds(g0, g1, z, z, z8, 0);
#else
  __builtin_amdgcn_tensor_load_to_lds(g0, g1, z, z, 0);
#endif
}
#endif

// ---------------------------------------------------------------------------
// prep kernels
// ---------------------------------------------------------------------------

// tiled transpose + f32->bf16: dst[g][n][k] = src[g][k][n]; grid (32,32,4)
__global__ __launch_bounds__(256) void k_transpose_cvt(
    const float* __restrict__ src, __bf16* __restrict__ dst) {
  __shared__ float tile[32][33];
  const int g  = blockIdx.z;
  const int kb = blockIdx.x * 32, nb = blockIdx.y * 32;
  const int tid = threadIdx.x;
  const int r = tid >> 3, c4 = (tid & 7) * 4;
  float4 v = *(const float4*)(src + ((size_t)g * K_ + kb + r) * H_ + nb + c4);
  tile[r][c4 + 0] = v.x; tile[r][c4 + 1] = v.y;
  tile[r][c4 + 2] = v.z; tile[r][c4 + 3] = v.w;
  __syncthreads();
  // write row n = nb + r, cols k = kb + c4..c4+3
  unsigned short s0 = __builtin_bit_cast(unsigned short, f2bf(tile[c4 + 0][r]));
  unsigned short s1 = __builtin_bit_cast(unsigned short, f2bf(tile[c4 + 1][r]));
  unsigned short s2 = __builtin_bit_cast(unsigned short, f2bf(tile[c4 + 2][r]));
  unsigned short s3 = __builtin_bit_cast(unsigned short, f2bf(tile[c4 + 3][r]));
  uint2 w; w.x = (unsigned)s0 | ((unsigned)s1 << 16);
  w.y = (unsigned)s2 | ((unsigned)s3 << 16);
  *(uint2*)(dst + ((size_t)g * H_ + nb + r) * K_ + kb + c4) = w;
}

// x f32 -> bf16, same layout; 4 elements/thread
__global__ void k_prep_x(const float* __restrict__ src, __bf16* __restrict__ dst, int n4) {
  int i = blockIdx.x * blockDim.x + threadIdx.x;
  if (i >= n4) return;
  float4 v = *(const float4*)(src + (size_t)i * 4);
  unsigned short s0 = __builtin_bit_cast(unsigned short, f2bf(v.x));
  unsigned short s1 = __builtin_bit_cast(unsigned short, f2bf(v.y));
  unsigned short s2 = __builtin_bit_cast(unsigned short, f2bf(v.z));
  unsigned short s3 = __builtin_bit_cast(unsigned short, f2bf(v.w));
  uint2 w; w.x = (unsigned)s0 | ((unsigned)s1 << 16);
  w.y = (unsigned)s2 | ((unsigned)s3 << 16);
  *(uint2*)(dst + (size_t)i * 4) = w;
}

__global__ void k_init(float* __restrict__ c0, __bf16* __restrict__ h0, int n) {
  int i = blockIdx.x * blockDim.x + threadIdx.x;
  if (i < n) { c0[i] = 0.f; h0[i] = __builtin_bit_cast(__bf16, (unsigned short)0); }
}

// ---------------------------------------------------------------------------
// Phase 1: xz[m][col] = x[m,:] @ W[col/H][:,col%H] + bias[col]
// grid (800, 32); 8 waves (2x4); block tile 64x128; wave tile 32x32
// ---------------------------------------------------------------------------
__global__ __launch_bounds__(256) void k_xproj(
    const __bf16* __restrict__ xbf,  // (51200, 1024)
    const __bf16* __restrict__ Wt,   // (4, 1024n, 1024k)  k-contiguous
    const float*  __restrict__ bias, // (4096,)
    float* __restrict__ xz)          // (51200, 4096)
{
  constexpr int MT = 64, NT = 128, KC = 32, NKC = K_ / KC;
  __shared__ __bf16 As[2][MT][KC + 8];
  __shared__ __bf16 Bs[2][NT][KC + 8];   // n-major

  const int tid  = threadIdx.x;
  const int lane = tid & 31;
  const int wave = tid >> 5;
  const int wm   = wave & 1;
  const int wn   = wave >> 1;
  const int mBase = blockIdx.x * MT;
  const int nGlob = blockIdx.y * NT;       // 0..4095, fixed gate per block
  const int l15   = lane & 15;
  const int half8 = (lane >> 4) * 8;

  v8f acc[2][2] = {};

#if USE_TDM
  if (wave == 0)
    tdm_load_2d((unsigned)(uintptr_t)&As[0][0][0], xbf + (size_t)mBase * K_, MT);
  else if (wave == 1)
    tdm_load_2d((unsigned)(uintptr_t)&Bs[0][0][0], Wt + (size_t)nGlob * K_, NT);

  for (int kc = 0; kc < NKC; ++kc) {
    const int cur = kc & 1;
    if (wave <= 1) __builtin_amdgcn_s_wait_tensorcnt(0);
    __syncthreads();                       // buf[cur] ready, buf[1-cur] free
    if (kc + 1 < NKC) {                    // prefetch next chunk under the math
      const int nxt = cur ^ 1, k1 = (kc + 1) * KC;
      if (wave == 0)
        tdm_load_2d((unsigned)(uintptr_t)&As[nxt][0][0],
                    xbf + (size_t)mBase * K_ + k1, MT);
      else if (wave == 1)
        tdm_load_2d((unsigned)(uintptr_t)&Bs[nxt][0][0],
                    Wt + (size_t)nGlob * K_ + k1, NT);
    }
    v16bf af[2], bq[2];
    #pragma unroll
    for (int i = 0; i < 2; ++i) {
      int m = wm * 32 + i * 16 + l15;
      #pragma unroll
      for (int e = 0; e < 8; ++e) {
        af[i][e]     = As[cur][m][half8 + e];
        af[i][e + 8] = As[cur][m][half8 + e + 16];
      }
    }
    #pragma unroll
    for (int j = 0; j < 2; ++j) {
      int n = wn * 32 + j * 16 + l15;
      #pragma unroll
      for (int e = 0; e < 8; ++e) {
        bq[j][e]     = Bs[cur][n][half8 + e];
        bq[j][e + 8] = Bs[cur][n][half8 + e + 16];
      }
    }
    #pragma unroll
    for (int i = 0; i < 2; ++i)
      #pragma unroll
      for (int j = 0; j < 2; ++j)
        acc[i][j] = __builtin_amdgcn_wmma_f32_16x16x32_bf16(
            false, af[i], false, bq[j], (short)0, acc[i][j], false, false);
  }
#else
  for (int k0 = 0; k0 < K_; k0 += KC) {
    #pragma unroll
    for (int it = 0; it < 2; ++it) {
      int idx = tid + it * 256, row = idx >> 3, cg = (idx & 7) * 4;
      *(uint2*)&As[0][row][cg] = *(const uint2*)(xbf + (size_t)(mBase + row) * K_ + k0 + cg);
    }
    #pragma unroll
    for (int it = 0; it < 4; ++it) {
      int idx = tid + it * 256, row = idx >> 3, cg = (idx & 7) * 4;
      *(uint2*)&Bs[0][row][cg] = *(const uint2*)(Wt + (size_t)(nGlob + row) * K_ + k0 + cg);
    }
    __syncthreads();
    v16bf af[2], bq[2];
    #pragma unroll
    for (int i = 0; i < 2; ++i) {
      int m = wm * 32 + i * 16 + l15;
      #pragma unroll
      for (int e = 0; e < 8; ++e) {
        af[i][e]     = As[0][m][half8 + e];
        af[i][e + 8] = As[0][m][half8 + e + 16];
      }
    }
    #pragma unroll
    for (int j = 0; j < 2; ++j) {
      int n = wn * 32 + j * 16 + l15;
      #pragma unroll
      for (int e = 0; e < 8; ++e) {
        bq[j][e]     = Bs[0][n][half8 + e];
        bq[j][e + 8] = Bs[0][n][half8 + e + 16];
      }
    }
    #pragma unroll
    for (int i = 0; i < 2; ++i)
      #pragma unroll
      for (int j = 0; j < 2; ++j)
        acc[i][j] = __builtin_amdgcn_wmma_f32_16x16x32_bf16(
            false, af[i], false, bq[j], (short)0, acc[i][j], false, false);
    __syncthreads();
  }
#endif

  #pragma unroll
  for (int i = 0; i < 2; ++i) {
    #pragma unroll
    for (int j = 0; j < 2; ++j) {
      int col = nGlob + wn * 32 + j * 16 + l15;
      float bv = bias[col];
      #pragma unroll
      for (int e = 0; e < 8; ++e) {
        int m = mBase + wm * 32 + i * 16 + half8 + e;
        xz[(size_t)m * N4_ + col] = acc[i][j][e] + bv;
      }
    }
  }
}

// ---------------------------------------------------------------------------
// Phase 2: one time step. grid (16,16) = (H/64, B/32); 8 waves (2x4);
// wave computes one 16x16 tile for all 4 gates (gate math stays wave-local).
// ---------------------------------------------------------------------------
__global__ __launch_bounds__(256) void k_step(
    const __bf16* __restrict__ hprev,  // (B, H) bf16
    const __bf16* __restrict__ Ut,     // (4, 1024n, 1024k) k-contiguous
    const float*  __restrict__ xz,     // (51200, 4096), row = b*T + t
    float*        __restrict__ c_st,   // (B, H)
    float*        __restrict__ hseq,   // (B, T, H)
    __bf16*       __restrict__ hnext,  // (B, H)
    int t)
{
  constexpr int MT = 32, NT = 64, KC = 32, NKC = K_ / KC;
  __shared__ __bf16 As[2][MT][KC + 8];
  __shared__ __bf16 Bs[2][4][NT][KC + 8];   // n-major per gate

  const int tid  = threadIdx.x;
  const int lane = tid & 31;
  const int wave = tid >> 5;
  const int wm   = wave & 1;
  const int wn   = wave >> 1;
  const int mBase = blockIdx.y * MT;
  const int nBase = blockIdx.x * NT;
  const int l15   = lane & 15;
  const int half8 = (lane >> 4) * 8;

  v8f acc[4];
  {
    int n = nBase + wn * 16 + l15;
    #pragma unroll
    for (int gg = 0; gg < 4; ++gg) {
      #pragma unroll
      for (int e = 0; e < 8; ++e) {
        int m = mBase + wm * 16 + half8 + e;
        acc[gg][e] = xz[((size_t)m * T_ + t) * N4_ + gg * H_ + n];
      }
    }
  }

#if USE_TDM
  if (wave == 0)
    tdm_load_2d((unsigned)(uintptr_t)&As[0][0][0], hprev + (size_t)mBase * H_, MT);
  else if (wave <= 4)
    tdm_load_2d((unsigned)(uintptr_t)&Bs[0][wave - 1][0][0],
                Ut + ((size_t)(wave - 1) * H_ + nBase) * K_, NT);

  for (int kc = 0; kc < NKC; ++kc) {
    const int cur = kc & 1;
    if (wave <= 4) __builtin_amdgcn_s_wait_tensorcnt(0);
    __syncthreads();                       // buf[cur] ready, buf[1-cur] free
    if (kc + 1 < NKC) {                    // prefetch next chunk under the math
      const int nxt = cur ^ 1, k1 = (kc + 1) * KC;
      if (wave == 0)
        tdm_load_2d((unsigned)(uintptr_t)&As[nxt][0][0],
                    hprev + (size_t)mBase * H_ + k1, MT);
      else if (wave <= 4)
        tdm_load_2d((unsigned)(uintptr_t)&Bs[nxt][wave - 1][0][0],
                    Ut + ((size_t)(wave - 1) * H_ + nBase) * K_ + k1, NT);
    }
    v16bf af;
    {
      int m = wm * 16 + l15;
      #pragma unroll
      for (int e = 0; e < 8; ++e) {
        af[e]     = As[cur][m][half8 + e];
        af[e + 8] = As[cur][m][half8 + e + 16];
      }
    }
    #pragma unroll
    for (int gg = 0; gg < 4; ++gg) {
      v16bf bq;
      int n = wn * 16 + l15;
      #pragma unroll
      for (int e = 0; e < 8; ++e) {
        bq[e]     = Bs[cur][gg][n][half8 + e];
        bq[e + 8] = Bs[cur][gg][n][half8 + e + 16];
      }
      acc[gg] = __builtin_amdgcn_wmma_f32_16x16x32_bf16(
          false, af, false, bq, (short)0, acc[gg], false, false);
    }
  }
#else
  for (int k0 = 0; k0 < K_; k0 += KC) {
    {
      int row = tid >> 3, cg = (tid & 7) * 4;
      *(uint2*)&As[0][row][cg] = *(const uint2*)(hprev + (size_t)(mBase + row) * H_ + k0 + cg);
    }
    #pragma unroll
    for (int it = 0; it < 8; ++it) {
      int idx = tid + it * 256;            // 0..2047
      int gg = idx >> 9, rem = idx & 511;
      int row = rem >> 3, cg = (rem & 7) * 4;
      *(uint2*)&Bs[0][gg][row][cg] =
          *(const uint2*)(Ut + ((size_t)gg * H_ + nBase + row) * K_ + k0 + cg);
    }
    __syncthreads();
    v16bf af;
    {
      int m = wm * 16 + l15;
      #pragma unroll
      for (int e = 0; e < 8; ++e) {
        af[e]     = As[0][m][half8 + e];
        af[e + 8] = As[0][m][half8 + e + 16];
      }
    }
    #pragma unroll
    for (int gg = 0; gg < 4; ++gg) {
      v16bf bq;
      int n = wn * 16 + l15;
      #pragma unroll
      for (int e = 0; e < 8; ++e) {
        bq[e]     = Bs[0][gg][n][half8 + e];
        bq[e + 8] = Bs[0][gg][n][half8 + e + 16];
      }
      acc[gg] = __builtin_amdgcn_wmma_f32_16x16x32_bf16(
          false, af, false, bq, (short)0, acc[gg], false, false);
    }
    __syncthreads();
  }
#endif

  {
    int n = nBase + wn * 16 + l15;
    #pragma unroll
    for (int e = 0; e < 8; ++e) {
      int m = mBase + wm * 16 + half8 + e;
      float ig = sigmoid_f(acc[0][e]);
      float fg = sigmoid_f(acc[1][e]);
      float og = sigmoid_f(acc[2][e]);
      float ct = b2bsqrt_f(acc[3][e]);
      size_t idx = (size_t)m * H_ + n;
      float cv = fg * c_st[idx] + ig * ct;
      c_st[idx] = cv;
      float hv = og * b2bsqrt_f(cv);
      hseq[((size_t)m * T_ + t) * H_ + n] = hv;
      hnext[idx] = f2bf(hv);
    }
  }
}

// ---------------------------------------------------------------------------
// Phase 3: LayerNorm + FC(1024 -> 10). One block per (b,t) row.
// ---------------------------------------------------------------------------
__global__ __launch_bounds__(256) void k_lnfc(
    const float* __restrict__ hseq,
    const float* __restrict__ ln_g,
    const float* __restrict__ ln_b,
    const float* __restrict__ fc_w,  // (1024, 10)
    const float* __restrict__ fc_b,  // (10,)
    float* __restrict__ out)         // (B, T, 10)
{
  const int r   = blockIdx.x;
  const int tid = threadIdx.x;
  __shared__ float rs[256], rss[256];
  __shared__ float lg[256][11];

  const float* row = hseq + (size_t)r * H_;
  float4 v4 = *(const float4*)(row + tid * 4);
  float v[4] = {v4.x, v4.y, v4.z, v4.w};
  float s = 0.f, ss = 0.f;
  #pragma unroll
  for (int e = 0; e < 4; ++e) { s += v[e]; ss += v[e] * v[e]; }
  rs[tid] = s; rss[tid] = ss;
  __syncthreads();
  for (int st = 128; st > 0; st >>= 1) {
    if (tid < st) { rs[tid] += rs[tid + st]; rss[tid] += rss[tid + st]; }
    __syncthreads();
  }
  float mu  = rs[0] * (1.f / H_);
  float var = rss[0] * (1.f / H_) - mu * mu;
  float inv = rsqrtf(var + 1e-5f);

  float part[10];
  #pragma unroll
  for (int c = 0; c < 10; ++c) part[c] = 0.f;
  #pragma unroll
  for (int e = 0; e < 4; ++e) {
    int h = tid * 4 + e;
    float y = (v[e] - mu) * inv * ln_g[h] + ln_b[h];
    const float* w = fc_w + h * 10;
    #pragma unroll
    for (int c = 0; c < 10; ++c) part[c] += y * w[c];
  }
  #pragma unroll
  for (int c = 0; c < 10; ++c) lg[tid][c] = part[c];
  __syncthreads();
  for (int st = 128; st > 0; st >>= 1) {
    if (tid < st)
      #pragma unroll
      for (int c = 0; c < 10; ++c) lg[tid][c] += lg[tid + st][c];
    __syncthreads();
  }
  if (tid < 10) out[(size_t)r * 10 + tid] = lg[0][tid] + fc_b[tid];
}

// ---------------------------------------------------------------------------
extern "C" void kernel_launch(void* const* d_in, const int* in_sizes, int n_in,
                              void* d_out, int out_size, void* d_ws, size_t ws_size,
                              hipStream_t stream) {
  const float* x    = (const float*)d_in[0];
  const float* W    = (const float*)d_in[1];
  const float* U    = (const float*)d_in[2];
  const float* bias = (const float*)d_in[3];
  const float* ln_g = (const float*)d_in[4];
  const float* ln_b = (const float*)d_in[5];
  const float* fc_w = (const float*)d_in[6];
  const float* fc_b = (const float*)d_in[7];
  float* out = (float*)d_out;

  char* ws = (char*)d_ws;
  size_t off = 0;
  auto carve = [&](size_t bytes) -> void* {
    void* p = ws + off;
    off += (bytes + 255) & ~(size_t)255;
    return p;
  };
  __bf16* Wt   = (__bf16*)carve((size_t)4 * K_ * H_ * 2);          //   8 MB
  __bf16* Ut   = (__bf16*)carve((size_t)4 * K_ * H_ * 2);          //   8 MB
  __bf16* xbf  = (__bf16*)carve((size_t)B_ * T_ * K_ * 2);         // 105 MB
  float*  xz   = (float*) carve((size_t)B_ * T_ * N4_ * 4);        // 839 MB
  float*  hseq = (float*) carve((size_t)B_ * T_ * H_ * 4);         // 210 MB
  float*  cst  = (float*) carve((size_t)B_ * H_ * 4);              //   2 MB
  __bf16* hA   = (__bf16*)carve((size_t)B_ * H_ * 2);              //   1 MB
  __bf16* hB   = (__bf16*)carve((size_t)B_ * H_ * 2);              //   1 MB
  (void)ws_size; (void)in_sizes; (void)n_in; (void)out_size;

  dim3 gt(K_ / 32, H_ / 32, 4);
  k_transpose_cvt<<<gt, 256, 0, stream>>>(W, Wt);
  k_transpose_cvt<<<gt, 256, 0, stream>>>(U, Ut);
  k_prep_x<<<(B_ * T_ * K_ / 4 + 255) / 256, 256, 0, stream>>>(x, xbf, B_ * T_ * K_ / 4);
  k_init<<<(B_ * H_ + 255) / 256, 256, 0, stream>>>(cst, hA, B_ * H_);

  dim3 g1(B_ * T_ / 64, N4_ / 128);           // (800, 32)
  k_xproj<<<g1, 256, 0, stream>>>(xbf, Wt, bias, xz);

  dim3 g2(H_ / 64, B_ / 32);                  // (16, 16)
  __bf16* hin = hA; __bf16* hout = hB;
  for (int t = 0; t < T_; ++t) {
    k_step<<<g2, 256, 0, stream>>>(hin, Ut, xz, cst, hseq, hout, t);
    __bf16* tmp = hin; hin = hout; hout = tmp;
  }

  k_lnfc<<<B_ * T_, 256, 0, stream>>>(hseq, ln_g, ln_b, fc_w, fc_b, out);
}
